// CustomInformerEncoderLayer_31447750541922
// MI455X (gfx1250) — matrix operand order repaired and checked by reading, not compiled, not run on hardware
//
#include <hip/hip_runtime.h>
#include <math.h>
#include <stdint.h>

typedef _Float16 half_t;
typedef __attribute__((ext_vector_type(16))) _Float16 v16h;
typedef __attribute__((ext_vector_type(8)))  float    v8f;
typedef int v4i_vec __attribute__((vector_size(16)));

// Problem constants (match reference)
constexpr int Bb  = 8;
constexpr int Ss  = 1024;
constexpr int Dd  = 1024;
constexpr int Hh  = 16;
constexpr int HDd = 64;
constexpr int Ff  = 4096;
constexpr int Mm  = Bb * Ss;           // 8192 rows
constexpr float SCALING = 0.125f;      // 64^-0.5
constexpr float EPS = 1e-5f;

// ---------------------------------------------------------------------------
// CDNA5 async global->LDS copy (ASYNCcnt-tracked), with portable fallback
// ---------------------------------------------------------------------------
#if defined(__AMDGCN__) && __has_builtin(__builtin_amdgcn_global_load_async_to_lds_b128)
#define ASYNC_LDS 1
#else
#define ASYNC_LDS 0
#endif

__device__ __forceinline__ void async_copy16(const half_t* g, half_t* l) {
#if ASYNC_LDS
    // Builtin signature (from clang diagnostic): (v4i AS1*, v4i AS3*, Ii, Ii).
    // AS1 pointer: identical bit pattern; AS3 pointer: flat LDS addr[31:0].
    __builtin_amdgcn_global_load_async_to_lds_b128(
        (__attribute__((address_space(1))) v4i_vec*)(uintptr_t)g,
        (__attribute__((address_space(3))) v4i_vec*)(uint32_t)(uintptr_t)l,
        0, 0);
#else
    *(float4*)l = *(const float4*)g;
#endif
}

template<int N>
__device__ __forceinline__ void wait_async() {
#if ASYNC_LDS
#if __has_builtin(__builtin_amdgcn_s_wait_asynccnt)
    __builtin_amdgcn_s_wait_asynccnt(N);
#else
    asm volatile("s_wait_asynccnt %0" :: "i"(N) : "memory");
#endif
#endif
}

__device__ __forceinline__ float gelu_exact(float x) {
    return 0.5f * x * (1.0f + erff(x * 0.70710678118654752f));
}

// ---------------------------------------------------------------------------
// f32 -> f16 conversion (grid-stride)
// ---------------------------------------------------------------------------
__global__ void cvt_f32_to_f16_kernel(const float* __restrict__ src,
                                      half_t* __restrict__ dst, size_t n) {
    size_t i = (size_t)blockIdx.x * blockDim.x + threadIdx.x;
    size_t stride = (size_t)gridDim.x * blockDim.x;
    for (; i < n; i += stride) dst[i] = (half_t)src[i];
}

// ---------------------------------------------------------------------------
// GEMM: C[M,N] = act((A[M,K] @ W[N,K]^T + bias[N]) * scale)
// Block tile 128x128, BK=32, double-buffered LDS fed by async copies.
// 256 threads = 8 waves arranged 4(m) x 2(n); each wave owns 32x64 =
// 2x4 WMMA 16x16x32 f16 tiles (8 WMMAs per k-step), f32 accumulate.
// ---------------------------------------------------------------------------
template<int ACT>
__global__ __launch_bounds__(256)
void gemm_bias_kernel(const half_t* __restrict__ A, const half_t* __restrict__ W,
                      const float* __restrict__ bias, float scale,
                      int M, int N, int K,
                      half_t* __restrict__ out16, float* __restrict__ out32)
{
    constexpr int BM = 128, BN = 128, BK = 32, BKP = BK + 8; // +8 halfs: bank-spread, 16B aligned
    __shared__ __align__(16) half_t As[2][BM][BKP];
    __shared__ __align__(16) half_t Ws[2][BN][BKP];

    const int tid  = threadIdx.x;
    const int lane = tid & 31;
    const int wv   = tid >> 5;      // wave 0..7
    const int wm   = wv >> 1;       // 0..3  (32-row group)
    const int wn   = wv & 1;        // 0..1  (64-col group)
    const int hs   = lane >> 4;     // lane half
    const int l15  = lane & 15;
    const int m0   = blockIdx.y * BM;
    const int n0   = blockIdx.x * BN;

    v8f acc[2][4];
    #pragma unroll
    for (int i = 0; i < 2; ++i)
        #pragma unroll
        for (int j = 0; j < 4; ++j) {
            v8f z = {0.f,0.f,0.f,0.f,0.f,0.f,0.f,0.f};
            acc[i][j] = z;
        }

    // Stage one BK-slice of A (128x32) and W (128x32): 2+2 16B chunks/thread.
    auto stage = [&](int buf, int k0) {
        #pragma unroll
        for (int i = 0; i < 2; ++i) {
            int c = tid + i * 256;          // 0..511 over 128x32 halfs
            int r = c >> 2, col = (c & 3) * 8;
            async_copy16(&A[(size_t)(m0 + r) * K + k0 + col], &As[buf][r][col]);
        }
        #pragma unroll
        for (int i = 0; i < 2; ++i) {
            int c = tid + i * 256;
            int r = c >> 2, col = (c & 3) * 8;
            async_copy16(&W[(size_t)(n0 + r) * K + k0 + col], &Ws[buf][r][col]);
        }
    };

    const int nk = K / BK;
    stage(0, 0);
    for (int s = 0; s < nk; ++s) {
        const int cur = s & 1;
        if (s + 1 < nk) {
            stage(cur ^ 1, (s + 1) * BK);   // issue next tile (4 async/thread)
            wait_async<4>();                // current tile's 4 complete
        } else {
            wait_async<0>();
        }
        __syncthreads();

        #pragma unroll
        for (int tm = 0; tm < 2; ++tm) {
            // A fragment: lane = M row, K split {hs*8..+7, 16+hs*8..+7}
            const half_t* ap = &As[cur][wm * 32 + tm * 16 + l15][0];
            v16h a;
            #pragma unroll
            for (int i = 0; i < 8; ++i) {
                a[i]     = ap[hs * 8 + i];
                a[8 + i] = ap[16 + hs * 8 + i];
            }
            #pragma unroll
            for (int tn = 0; tn < 4; ++tn) {
                // B fragment of W^T: lane = N col, K split {hs*8..+7, 16+hs*8..+7}
                const half_t* bp = &Ws[cur][wn * 64 + tn * 16 + l15][0];
                v16h b;
                #pragma unroll
                for (int i = 0; i < 16; ++i) b[i] = bp[(i < 8) ? hs * 8 + i : 16 + hs * 8 + (i - 8)];
                acc[tm][tn] = __builtin_amdgcn_wmma_f32_16x16x32_f16(
                    false, a, false, b, (short)0, acc[tm][tn], false, false);
            }
        }
        __syncthreads();
    }

    // Epilogue: C layout -> VGPR r, lane: m = r + 8*hs, n = l15
    #pragma unroll
    for (int tm = 0; tm < 2; ++tm)
        #pragma unroll
        for (int tn = 0; tn < 4; ++tn)
            #pragma unroll
            for (int r = 0; r < 8; ++r) {
                int row = m0 + wm * 32 + tm * 16 + r + 8 * hs;
                int col = n0 + wn * 64 + tn * 16 + l15;
                float vv = (acc[tm][tn][r] + bias[col]) * scale;
                if (ACT == 1) vv = gelu_exact(vv);
                size_t idx = (size_t)row * N + col;
                if (out32) out32[idx] = vv;
                if (out16) out16[idx] = (half_t)vv;
            }
}

// ---------------------------------------------------------------------------
// Flash attention: block = (64 queries, one head, one batch), 128 threads.
// Each wave owns 16 query rows. Online softmax, f32 running stats.
// q is pre-scaled by SCALING in the Q projection epilogue.
// ---------------------------------------------------------------------------
__global__ __launch_bounds__(128)
void flash_attn_kernel(const half_t* __restrict__ q, const half_t* __restrict__ k,
                       const half_t* __restrict__ v, half_t* __restrict__ o)
{
    constexpr int LP = 72;  // 64 + 8 pad halfs (144B rows, 16B aligned)
    __shared__ __align__(16) half_t Kt[64][LP];
    __shared__ __align__(16) half_t Vt[64][LP];
    __shared__ __align__(16) half_t Pt[64][LP];

    const int tid  = threadIdx.x;
    const int lane = tid & 31;
    const int wv   = tid >> 5;      // wave 0..3: query rows [wv*16, wv*16+16)
    const int hs   = lane >> 4;
    const int l15  = lane & 15;
    const int q0   = blockIdx.x * 64;
    const int h    = blockIdx.y;
    const int bz   = blockIdx.z;
    const size_t headBase = ((size_t)bz * Ss) * Dd + (size_t)h * HDd;

    // Q fragments (2 K-chunks of 32 over HD=64), kept in registers
    v16h qf[2];
    {
        const half_t* qrow = q + headBase + (size_t)(q0 + wv * 16 + l15) * Dd;
        #pragma unroll
        for (int f = 0; f < 2; ++f)
            #pragma unroll
            for (int i = 0; i < 8; ++i) {
                qf[f][i]     = qrow[f * 32 + hs * 8 + i];
                qf[f][8 + i] = qrow[f * 32 + 16 + hs * 8 + i];
            }
    }

    v8f Ot[4];
    #pragma unroll
    for (int t = 0; t < 4; ++t) { v8f z = {0.f,0.f,0.f,0.f,0.f,0.f,0.f,0.f}; Ot[t] = z; }
    float mrow[8], lrow[8];
    #pragma unroll
    for (int r = 0; r < 8; ++r) { mrow[r] = -1e30f; lrow[r] = 0.f; }

    for (int kv0 = 0; kv0 < Ss; kv0 += 64) {
        // Async-stage K,V tiles: 64x64 halfs = 512 chunks each, 4+4 per thread
        #pragma unroll
        for (int c = tid; c < 512; c += 128) {
            int r = c >> 3, col = (c & 7) * 8;
            async_copy16(&k[headBase + (size_t)(kv0 + r) * Dd + col], &Kt[r][col]);
            async_copy16(&v[headBase + (size_t)(kv0 + r) * Dd + col], &Vt[r][col]);
        }
        if (kv0 + 64 < Ss)   // warm L2 for next KV tile (global_prefetch_b8)
            __builtin_prefetch(&k[headBase + (size_t)(kv0 + 64 + (tid & 63)) * Dd], 0, 1);
        wait_async<0>();
        __syncthreads();

        // scores: S[16 rows x 64 kv] = Q @ K^T, K-dim = HD = 64 (2 chained WMMA)
        v8f Sc[4];
        #pragma unroll
        for (int nt = 0; nt < 4; ++nt) {
            v8f s = {0.f,0.f,0.f,0.f,0.f,0.f,0.f,0.f};
            #pragma unroll
            for (int f = 0; f < 2; ++f) {
                // B fragment of K^T: lane = kv col -> HD split {hs*8..+7, 16+hs*8..+7} of K row
                const half_t* bp = &Kt[nt * 16 + l15][f * 32];
                v16h b;
                #pragma unroll
                for (int i = 0; i < 16; ++i) b[i] = bp[(i < 8) ? hs * 8 + i : 16 + hs * 8 + (i - 8)];
                s = __builtin_amdgcn_wmma_f32_16x16x32_f16(
                    false, qf[f], false, b, (short)0, s, false, false);
            }
            Sc[nt] = s;
        }

        // online softmax: per-row max/sum across 64 kv columns
        #pragma unroll
        for (int r = 0; r < 8; ++r) {
            float vmax = -1e30f;
            #pragma unroll
            for (int nt = 0; nt < 4; ++nt) vmax = fmaxf(vmax, Sc[nt][r]);
            #pragma unroll
            for (int off = 1; off < 16; off <<= 1)
                vmax = fmaxf(vmax, __shfl_xor(vmax, off, 32));
            float mnew = fmaxf(mrow[r], vmax);
            float alpha = __expf(mrow[r] - mnew);
            mrow[r] = mnew;
            lrow[r] *= alpha;
            #pragma unroll
            for (int t = 0; t < 4; ++t) Ot[t][r] *= alpha;
            float ps = 0.f;
            #pragma unroll
            for (int nt = 0; nt < 4; ++nt) {
                float p = __expf(Sc[nt][r] - mnew);
                Sc[nt][r] = p;
                ps += p;
            }
            #pragma unroll
            for (int off = 1; off < 16; off <<= 1)
                ps += __shfl_xor(ps, off, 32);
            lrow[r] += ps;
        }

        // write P strip to LDS (C layout -> row-major) for A-fragment reload
        #pragma unroll
        for (int nt = 0; nt < 4; ++nt)
            #pragma unroll
            for (int r = 0; r < 8; ++r)
                Pt[wv * 16 + r + 8 * hs][nt * 16 + l15] = (half_t)Sc[nt][r];
        __syncthreads();

        // O += P @ V, K-dim = 64 kv (2 chained WMMA per 16-wide HD tile)
        #pragma unroll
        for (int f = 0; f < 2; ++f) {
            const half_t* pp = &Pt[wv * 16 + l15][0];
            v16h a;
            #pragma unroll
            for (int i = 0; i < 8; ++i) {
                a[i]     = pp[f * 32 + hs * 8 + i];
                a[8 + i] = pp[f * 32 + 16 + hs * 8 + i];
            }
            #pragma unroll
            for (int ht = 0; ht < 4; ++ht) {
                v16h b;  // B fragment of V: lane = hd col, strided kv reads
                #pragma unroll
                for (int i = 0; i < 16; ++i)
                    b[i] = Vt[f * 32 + ((i < 8) ? hs * 8 + i : 16 + hs * 8 + (i - 8))][ht * 16 + l15];
                Ot[ht] = __builtin_amdgcn_wmma_f32_16x16x32_f16(
                    false, a, false, b, (short)0, Ot[ht], false, false);
            }
        }
        __syncthreads();
    }

    // normalize and store attention output in [B,S,D] layout
    #pragma unroll
    for (int ht = 0; ht < 4; ++ht)
        #pragma unroll
        for (int r = 0; r < 8; ++r) {
            int row = q0 + wv * 16 + r + 8 * hs;
            o[((size_t)bz * Ss + row) * Dd + h * HDd + ht * 16 + l15] =
                (half_t)(Ot[ht][r] / lrow[r]);
        }
}

// ---------------------------------------------------------------------------
// residual add + LayerNorm over last dim (D=1024), one block per row
// ---------------------------------------------------------------------------
__global__ __launch_bounds__(256)
void add_ln_kernel(const float* __restrict__ x, const float* __restrict__ res,
                   const float* __restrict__ g, const float* __restrict__ bta,
                   float* __restrict__ out32, half_t* __restrict__ out16)
{
    __shared__ float redS[8], redSS[8];
    __shared__ float s_mean, s_rstd;
    const int tid  = threadIdx.x;
    const int lane = tid & 31;
    const int wv   = tid >> 5;
    const size_t row = blockIdx.x;
    const float* px = x + row * Dd;
    const float* pr = res + row * Dd;

    float vals[4], s = 0.f, ss = 0.f;
    #pragma unroll
    for (int i = 0; i < 4; ++i) {
        int idx = tid + i * 256;
        float u = px[idx] + pr[idx];
        vals[i] = u; s += u; ss += u * u;
    }
    #pragma unroll
    for (int off = 1; off < 32; off <<= 1) {
        s  += __shfl_xor(s, off, 32);
        ss += __shfl_xor(ss, off, 32);
    }
    if (lane == 0) { redS[wv] = s; redSS[wv] = ss; }
    __syncthreads();
    if (tid == 0) {
        float S = 0.f, SS = 0.f;
        #pragma unroll
        for (int w = 0; w < 8; ++w) { S += redS[w]; SS += redSS[w]; }
        float mean = S * (1.0f / Dd);
        float var  = SS * (1.0f / Dd) - mean * mean;
        s_mean = mean;
        s_rstd = rsqrtf(var + EPS);
    }
    __syncthreads();
    const float mean = s_mean, rstd = s_rstd;
    #pragma unroll
    for (int i = 0; i < 4; ++i) {
        int idx = tid + i * 256;
        float y = (vals[i] - mean) * rstd * g[idx] + bta[idx];
        if (out32) out32[row * Dd + idx] = y;
        if (out16) out16[row * Dd + idx] = (half_t)y;
    }
}

// ---------------------------------------------------------------------------
// launch
// ---------------------------------------------------------------------------
extern "C" void kernel_launch(void* const* d_in, const int* in_sizes, int n_in,
                              void* d_out, int out_size, void* d_ws, size_t ws_size,
                              hipStream_t stream) {
    const float* hidden = (const float*)d_in[0];
    const float* wq     = (const float*)d_in[1];
    const float* bq     = (const float*)d_in[2];
    const float* wk     = (const float*)d_in[3];
    const float* bk     = (const float*)d_in[4];
    const float* wvp    = (const float*)d_in[5];
    const float* bv     = (const float*)d_in[6];
    const float* wo     = (const float*)d_in[7];
    const float* bo     = (const float*)d_in[8];
    const float* ln1_g  = (const float*)d_in[9];
    const float* ln1_b  = (const float*)d_in[10];
    const float* fc1_w  = (const float*)d_in[11];
    const float* fc1_b  = (const float*)d_in[12];
    const float* fc2_w  = (const float*)d_in[13];
    const float* fc2_b  = (const float*)d_in[14];
    const float* ln2_g  = (const float*)d_in[15];
    const float* ln2_b  = (const float*)d_in[16];

    char* ws = (char*)d_ws;
    size_t off = 0;
    auto alloc = [&](size_t bytes) { char* p = ws + off; off += (bytes + 255) & ~(size_t)255; return p; };

    half_t* xh    = (half_t*)alloc((size_t)Mm * Dd * 2);   // x in f16
    half_t* wqh   = (half_t*)alloc((size_t)Dd * Dd * 2);
    half_t* wkh   = (half_t*)alloc((size_t)Dd * Dd * 2);
    half_t* wvh   = (half_t*)alloc((size_t)Dd * Dd * 2);
    half_t* woh   = (half_t*)alloc((size_t)Dd * Dd * 2);
    half_t* fc1h  = (half_t*)alloc((size_t)Ff * Dd * 2);
    half_t* fc2h  = (half_t*)alloc((size_t)Dd * Ff * 2);
    half_t* qh    = (half_t*)alloc((size_t)Mm * Dd * 2);
    half_t* kh    = (half_t*)alloc((size_t)Mm * Dd * 2);
    half_t* vh    = (half_t*)alloc((size_t)Mm * Dd * 2);
    half_t* attnh = (half_t*)alloc((size_t)Mm * Dd * 2);
    float*  proj  = (float*) alloc((size_t)Mm * Dd * 4);
    float*  hs1   = (float*) alloc((size_t)Mm * Dd * 4);
    half_t* hs1h  = (half_t*)alloc((size_t)Mm * Dd * 2);
    half_t* ffh   = (half_t*)alloc((size_t)Mm * Ff * 2);
    float*  ff2   = (float*) alloc((size_t)Mm * Dd * 4);

    auto cvt = [&](const float* s, half_t* d, size_t n) {
        cvt_f32_to_f16_kernel<<<1024, 256, 0, stream>>>(s, d, n);
    };
    cvt(hidden, xh,   (size_t)Mm * Dd);
    cvt(wq,     wqh,  (size_t)Dd * Dd);
    cvt(wk,     wkh,  (size_t)Dd * Dd);
    cvt(wvp,    wvh,  (size_t)Dd * Dd);
    cvt(wo,     woh,  (size_t)Dd * Dd);
    cvt(fc1_w,  fc1h, (size_t)Ff * Dd);
    cvt(fc2_w,  fc2h, (size_t)Dd * Ff);

    dim3 blk(256);
    // Q/K/V projections (Q pre-scaled, as in reference)
    gemm_bias_kernel<0><<<dim3(Dd / 128, Mm / 128), blk, 0, stream>>>(
        xh, wqh, bq, SCALING, Mm, Dd, Dd, qh, nullptr);
    gemm_bias_kernel<0><<<dim3(Dd / 128, Mm / 128), blk, 0, stream>>>(
        xh, wkh, bk, 1.0f, Mm, Dd, Dd, kh, nullptr);
    gemm_bias_kernel<0><<<dim3(Dd / 128, Mm / 128), blk, 0, stream>>>(
        xh, wvh, bv, 1.0f, Mm, Dd, Dd, vh, nullptr);

    // flash attention
    flash_attn_kernel<<<dim3(Ss / 64, Hh, Bb), dim3(128), 0, stream>>>(qh, kh, vh, attnh);

    // output projection -> f32
    gemm_bias_kernel<0><<<dim3(Dd / 128, Mm / 128), blk, 0, stream>>>(
        attnh, woh, bo, 1.0f, Mm, Dd, Dd, nullptr, proj);

    // residual + LN1 -> hs1 (f32 residual) + hs1h (f16 for FFN)
    add_ln_kernel<<<Mm, blk, 0, stream>>>(proj, hidden, ln1_g, ln1_b, hs1, hs1h);

    // FFN: fc1 + exact GELU -> f16, then fc2 -> f32
    gemm_bias_kernel<1><<<dim3(Ff / 128, Mm / 128), blk, 0, stream>>>(
        hs1h, fc1h, fc1_b, 1.0f, Mm, Ff, Dd, ffh, nullptr);
    gemm_bias_kernel<0><<<dim3(Dd / 128, Mm / 128), blk, 0, stream>>>(
        ffh, fc2h, fc2_b, 1.0f, Mm, Dd, Ff, nullptr, ff2);

    // residual + LN2 -> final f32 output
    add_ln_kernel<<<Mm, blk, 0, stream>>>(ff2, hs1, ln2_g, ln2_b, (float*)d_out, nullptr);
}
